// MultiHeadAttentionBlock_17265768530410
// MI455X (gfx1250) — matrix operand order repaired and checked by reading, not compiled, not run on hardware
//
#include <hip/hip_runtime.h>

// ---------------------------------------------------------------------------
// MHA block for MI455X (gfx1250, wave32).  Pipeline:
//   1) f32 -> bf16 convert (inputs + weights)
//   2) Q/K/V projections:  Y = X @ W^T + b  via v_wmma_f32_16x16x32_bf16
//      (wave tile 32x64, prefetch on A streams)
//   3) flash attention (online softmax) via bf16 WMMA, f32 accum
//      (K tile staged with GLOBAL_LOAD_ASYNC_TO_LDS_B128)
// ---------------------------------------------------------------------------

#define SEQ  1024
#define EMB  1024
#define HEADS 16
#define HD   64
#define NB   4

typedef unsigned short u16;
typedef __attribute__((ext_vector_type(16))) __bf16 v16bf;
typedef __attribute__((ext_vector_type(8)))  float  v8f;
typedef __attribute__((ext_vector_type(8)))  u16    v8us;
typedef __attribute__((ext_vector_type(4)))  float  v4f;
typedef __attribute__((ext_vector_type(4)))  u16    v4us;
typedef __attribute__((ext_vector_type(4)))  int    v4i;

// Async global->LDS copy (gfx1250). Guarded: falls back to sync path if the
// toolchain does not declare the builtins.  Param types per clang diagnostic:
//   (v4i addrspace(1)*, v4i addrspace(3)*, imm offset, cpol)
#if defined(__HIP_DEVICE_COMPILE__) && defined(__gfx1250__) && \
    __has_builtin(__builtin_amdgcn_global_load_async_to_lds_b128) && \
    __has_builtin(__builtin_amdgcn_s_wait_asynccnt)
#define HAVE_ASYNC_LDS 1
typedef v4i __attribute__((address_space(1))) g_v4i;   // global int4
typedef v4i __attribute__((address_space(3))) s_v4i;   // LDS int4
#else
#define HAVE_ASYNC_LDS 0
#endif

union BFrag {
    v16bf bf;
    v8us  us[2];
};

__device__ __forceinline__ u16 f2bf(float f) {
    unsigned int u = __float_as_uint(f);
    u += 0x7FFFu + ((u >> 16) & 1u);   // round-to-nearest-even
    return (u16)(u >> 16);
}

__device__ __forceinline__ v8f bwmma(const BFrag& a, const BFrag& b, v8f c) {
    // D = A(16x32 bf16) x B(32x16 bf16) + C(16x16 f32)
    return __builtin_amdgcn_wmma_f32_16x16x32_bf16(
        /*neg_a=*/false, a.bf, /*neg_b=*/false, b.bf,
        /*c_mod=*/(short)0, c, /*reuse_a=*/false, /*reuse_b=*/false);
}

// ---------------------------------------------------------------------------
// Kernel 1: f32 -> bf16 (4 elements / thread)
// ---------------------------------------------------------------------------
__global__ void f32_to_bf16(const float* __restrict__ src, u16* __restrict__ dst, int n4) {
    int i = blockIdx.x * blockDim.x + threadIdx.x;
    if (i >= n4) return;
    v4f f = ((const v4f*)src)[i];
    v4us o;
    o.x = f2bf(f.x); o.y = f2bf(f.y); o.z = f2bf(f.z); o.w = f2bf(f.w);
    ((v4us*)dst)[i] = o;
}

// ---------------------------------------------------------------------------
// Kernel 2: projection GEMM  Y[M=4096,N=1024] = X[M,1024] @ W[N,1024]^T + b
// Block = 256 threads = 8 waves; block tile 256(M) x 64(N); wave tile 32x64.
// 8 WMMAs per k-step against 4 shared B-frag loads.
// ---------------------------------------------------------------------------
__global__ __launch_bounds__(256) void proj_gemm(
    const u16* __restrict__ X, const u16* __restrict__ W,
    const float* __restrict__ bias, u16* __restrict__ Y)
{
    const int lane  = threadIdx.x & 31;
    const int wave  = threadIdx.x >> 5;
    const int l16   = lane & 15;
    const int kbA   = (lane < 16) ? 0 : 8;    // A frag K-offset per lane half
    const int kbB   = (lane < 16) ? 0 : 8;    // B frag K-offset per lane half
    const int mBase = blockIdx.x * 256 + wave * 32;
    const int nBase = blockIdx.y * 64;

    v8f acc0[4] = {};   // rows mBase..mBase+15
    v8f acc1[4] = {};   // rows mBase+16..mBase+31

    const u16* xrow0 = X + (size_t)(mBase + l16) * EMB;
    const u16* xrow1 = X + (size_t)(mBase + 16 + l16) * EMB;
    const u16* wrow[4] = {
        W + (size_t)(nBase +  0 + l16) * EMB,
        W + (size_t)(nBase + 16 + l16) * EMB,
        W + (size_t)(nBase + 32 + l16) * EMB,
        W + (size_t)(nBase + 48 + l16) * EMB,
    };

    for (int k0 = 0; k0 < EMB; k0 += 32) {
        if (k0 + 128 < EMB) {                       // global_prefetch_b8
            __builtin_prefetch(xrow0 + k0 + 128, 0, 3);
            __builtin_prefetch(xrow1 + k0 + 128, 0, 3);
        }
        BFrag a0, a1;
        a0.us[0] = *(const v8us*)(xrow0 + k0 + kbA);        // K = kbA..kbA+7
        a0.us[1] = *(const v8us*)(xrow0 + k0 + kbA + 16);   // K = kbA+16..kbA+23
        a1.us[0] = *(const v8us*)(xrow1 + k0 + kbA);
        a1.us[1] = *(const v8us*)(xrow1 + k0 + kbA + 16);
#pragma unroll
        for (int j = 0; j < 4; ++j) {
            BFrag b;
            b.us[0] = *(const v8us*)(wrow[j] + k0 + kbB);      // K = kbB..kbB+7
            b.us[1] = *(const v8us*)(wrow[j] + k0 + kbB + 16); // K = kbB+16..kbB+23
            acc0[j] = bwmma(a0, b, acc0[j]);
            acc1[j] = bwmma(a1, b, acc1[j]);
        }
    }

    // epilogue: + bias, store bf16.  C layout: VGPR r -> row r (lanes 0-15) / r+8
#pragma unroll
    for (int j = 0; j < 4; ++j) {
        const int col = nBase + j * 16 + l16;
        const float bv = bias[col];
#pragma unroll
        for (int r = 0; r < 8; ++r) {
            const int rr = (lane < 16) ? r : r + 8;
            Y[(size_t)(mBase + rr) * EMB + col]      = f2bf(acc0[j][r] + bv);
            Y[(size_t)(mBase + 16 + rr) * EMB + col] = f2bf(acc1[j][r] + bv);
        }
    }
}

// ---------------------------------------------------------------------------
// Kernel 3: flash attention.
//   queries <- Vp (v projection), keys <- Kp, values <- Qp  (reference quirk)
// Block = 128 threads = 4 waves, one (n,h), 64 queries (16/wave).
// Key loop in blocks of 32; K-tile (async) + transposed V-tile staged in LDS.
// ---------------------------------------------------------------------------
#define KPAD 80   // u16 per sK row (160B, 16B-aligned)
#define VPAD 40   // u16 per sVT row (80B, 16B-aligned)

__global__ __launch_bounds__(128) void flash_attn(
    const u16* __restrict__ Qp,   // values role
    const u16* __restrict__ Kp,   // keys
    const u16* __restrict__ Vp,   // queries role
    float* __restrict__ out)
{
    __shared__ u16 sK[32][KPAD];     // [key][dim]   32 x 64 bf16
    __shared__ u16 sVT[64][VPAD];    // [dim][key]   transposed values tile
    __shared__ u16 sP[4][16][32];    // per-wave P re-layout buffer

    const int lane = threadIdx.x & 31;
    const int wave = threadIdx.x >> 5;
    const int l16  = lane & 15;
    const int kbA  = (lane < 16) ? 0 : 8;
    const int kbB  = (lane < 16) ? 0 : 8;

    const int nh = blockIdx.x;           // 0..63
    const int n  = nh >> 4;
    const int h  = nh & 15;
    const int qBase = blockIdx.y * 64 + wave * 16;

    // ---- Q fragments (head_dim 64 -> two K=32 frags), loaded once ----
    const u16* qrow = Vp + (size_t)(n * SEQ + qBase + l16) * EMB + h * HD;
    BFrag aQ0, aQ1;
    aQ0.us[0] = *(const v8us*)(qrow + kbA);
    aQ0.us[1] = *(const v8us*)(qrow + kbA + 16);
    aQ1.us[0] = *(const v8us*)(qrow + 32 + kbA);
    aQ1.us[1] = *(const v8us*)(qrow + 32 + kbA + 16);

    v8f o[4] = {};
    float mrow[8], lrow[8];
#pragma unroll
    for (int r = 0; r < 8; ++r) { mrow[r] = -1e30f; lrow[r] = 0.f; }

    const float sc = 0.03125f;   // 1/sqrt(1024)

    // cooperative loader mapping: 128 threads = 32 keys x 4 dim-chunks
    const int tkey = threadIdx.x & 31;
    const int tch  = threadIdx.x >> 5;

    for (int kb0 = 0; kb0 < SEQ; kb0 += 32) {
        __syncthreads();   // protect tiles from previous iteration's readers
        {
            const size_t rbase = (size_t)(n * SEQ + kb0 + tkey) * EMB + h * HD + tch * 16;
#if HAVE_ASYNC_LDS
            // K tile: direct global->LDS DMA, tracked by ASYNCcnt
            __builtin_amdgcn_global_load_async_to_lds_b128(
                (g_v4i*)(Kp + rbase),     (s_v4i*)&sK[tkey][tch * 16],     0, 0);
            __builtin_amdgcn_global_load_async_to_lds_b128(
                (g_v4i*)(Kp + rbase + 8), (s_v4i*)&sK[tkey][tch * 16 + 8], 0, 0);
#else
            v8us k0v = *(const v8us*)(Kp + rbase);
            v8us k1v = *(const v8us*)(Kp + rbase + 8);
            *(v8us*)(&sK[tkey][tch * 16])     = k0v;
            *(v8us*)(&sK[tkey][tch * 16 + 8]) = k1v;
#endif
            // V tile: needs transpose, so round-trip through VGPRs
            v8us v0v = *(const v8us*)(Qp + rbase);
            v8us v1v = *(const v8us*)(Qp + rbase + 8);
#pragma unroll
            for (int t = 0; t < 8; ++t) sVT[tch * 16 + t][tkey]     = v0v[t];
#pragma unroll
            for (int t = 0; t < 8; ++t) sVT[tch * 16 + 8 + t][tkey] = v1v[t];
        }
#if HAVE_ASYNC_LDS
        __builtin_amdgcn_s_wait_asynccnt(0);
#endif
        __syncthreads();

        // ---- scores S = Q . K^T  (2 x 16-key tiles, K-dim 64 = 2 WMMAs each)
        BFrag b;
        v8f s0 = {}, s1 = {};
        b.us[0] = *(const v8us*)(&sK[l16][kbB]);
        b.us[1] = *(const v8us*)(&sK[l16][kbB + 16]);
        s0 = bwmma(aQ0, b, s0);
        b.us[0] = *(const v8us*)(&sK[l16][32 + kbB]);
        b.us[1] = *(const v8us*)(&sK[l16][32 + kbB + 16]);
        s0 = bwmma(aQ1, b, s0);
        b.us[0] = *(const v8us*)(&sK[16 + l16][kbB]);
        b.us[1] = *(const v8us*)(&sK[16 + l16][kbB + 16]);
        s1 = bwmma(aQ0, b, s1);
        b.us[0] = *(const v8us*)(&sK[16 + l16][32 + kbB]);
        b.us[1] = *(const v8us*)(&sK[16 + l16][32 + kbB + 16]);
        s1 = bwmma(aQ1, b, s1);

        // ---- online softmax update (row = C-layout row per lane half) ----
#pragma unroll
        for (int r = 0; r < 8; ++r) {
            s0[r] *= sc; s1[r] *= sc;
            float v = fmaxf(s0[r], s1[r]);
            v = fmaxf(v, __shfl_xor(v, 1));   // xor 1/2/4/8 stay inside the
            v = fmaxf(v, __shfl_xor(v, 2));   // 16-lane half that owns the row
            v = fmaxf(v, __shfl_xor(v, 4));
            v = fmaxf(v, __shfl_xor(v, 8));
            const float mnew  = fmaxf(mrow[r], v);
            const float scale = __expf(mrow[r] - mnew);
            mrow[r] = mnew;
            const float p0 = __expf(s0[r] - mnew);
            const float p1 = __expf(s1[r] - mnew);
            s0[r] = p0; s1[r] = p1;
            float rs = p0 + p1;
            rs += __shfl_xor(rs, 1);
            rs += __shfl_xor(rs, 2);
            rs += __shfl_xor(rs, 4);
            rs += __shfl_xor(rs, 8);
            lrow[r] = lrow[r] * scale + rs;
            o[0][r] *= scale; o[1][r] *= scale; o[2][r] *= scale; o[3][r] *= scale;
        }

        // ---- re-layout P: C-format -> A-format via per-wave LDS ----
        {
            const int prow = (lane < 16) ? 0 : 8;
#pragma unroll
            for (int r = 0; r < 8; ++r) {
                sP[wave][prow + r][l16]      = f2bf(s0[r]);
                sP[wave][prow + r][l16 + 16] = f2bf(s1[r]);
            }
        }
        __syncthreads();

        BFrag aP;
        aP.us[0] = *(const v8us*)(&sP[wave][l16][kbA]);
        aP.us[1] = *(const v8us*)(&sP[wave][l16][kbA + 16]);

        // ---- O += P . V  (4 x 16-dim chunks, K = 32 keys) ----
#pragma unroll
        for (int j = 0; j < 4; ++j) {
            BFrag bv;
            bv.us[0] = *(const v8us*)(&sVT[j * 16 + l16][kbB]);
            bv.us[1] = *(const v8us*)(&sVT[j * 16 + l16][kbB + 16]);
            o[j] = bwmma(aP, bv, o[j]);
        }
    }

    // ---- normalize and store (f32 output) ----
    float inv[8];
#pragma unroll
    for (int r = 0; r < 8; ++r) inv[r] = 1.f / lrow[r];
#pragma unroll
    for (int j = 0; j < 4; ++j) {
        const int col = h * HD + j * 16 + l16;
#pragma unroll
        for (int r = 0; r < 8; ++r) {
            const int q = qBase + ((lane < 16) ? r : r + 8);
            out[(size_t)(n * SEQ + q) * EMB + col] = o[j][r] * inv[r];
        }
    }
}

// ---------------------------------------------------------------------------
// Host-side launch
// ---------------------------------------------------------------------------
extern "C" void kernel_launch(void* const* d_in, const int* in_sizes, int n_in,
                              void* d_out, int out_size, void* d_ws, size_t ws_size,
                              hipStream_t stream) {
    const float* qi = (const float*)d_in[0];
    const float* ki = (const float*)d_in[1];
    const float* vi = (const float*)d_in[2];
    const float* Wq = (const float*)d_in[3];
    const float* bq = (const float*)d_in[4];
    const float* Wk = (const float*)d_in[5];
    const float* bk = (const float*)d_in[6];
    const float* Wv = (const float*)d_in[7];
    const float* bv = (const float*)d_in[8];
    float* out = (float*)d_out;

    const size_t MK = (size_t)NB * SEQ * EMB;   // 4M activation elements
    const size_t NK = (size_t)EMB * EMB;        // 1M weight elements

    u16* ws = (u16*)d_ws;
    u16* xq = ws;            u16* xk = xq + MK;  u16* xv = xk + MK;
    u16* wq = xv + MK;       u16* wk = wq + NK;  u16* wv = wk + NK;
    u16* Qp = wv + NK;       u16* Kp = Qp + MK;  u16* Vp = Kp + MK;

    // 1) convert to bf16
    {
        const int tb = 256;
        int n4 = (int)(MK / 4);
        f32_to_bf16<<<(n4 + tb - 1) / tb, tb, 0, stream>>>(qi, xq, n4);
        f32_to_bf16<<<(n4 + tb - 1) / tb, tb, 0, stream>>>(ki, xk, n4);
        f32_to_bf16<<<(n4 + tb - 1) / tb, tb, 0, stream>>>(vi, xv, n4);
        int w4 = (int)(NK / 4);
        f32_to_bf16<<<(w4 + tb - 1) / tb, tb, 0, stream>>>(Wq, wq, w4);
        f32_to_bf16<<<(w4 + tb - 1) / tb, tb, 0, stream>>>(Wk, wk, w4);
        f32_to_bf16<<<(w4 + tb - 1) / tb, tb, 0, stream>>>(Wv, wv, w4);
    }

    // 2) projections (store bf16)
    {
        dim3 grid(NB * SEQ / 256, EMB / 64);
        proj_gemm<<<grid, 256, 0, stream>>>(xq, wq, bq, Qp);
        proj_gemm<<<grid, 256, 0, stream>>>(xk, wk, bk, Kp);
        proj_gemm<<<grid, 256, 0, stream>>>(xv, wv, bv, Vp);
    }

    // 3) attention: grid = (batch*heads, query blocks of 64)
    {
        dim3 grid(NB * HEADS, SEQ / 64);
        flash_attn<<<grid, 128, 0, stream>>>(Qp, Kp, Vp, out);
    }
}
